// SimpleRNNModel_26697516712384
// MI455X (gfx1250) — matrix operand order repaired and checked by reading, not compiled, not run on hardware
//
#include <hip/hip_runtime.h>

#define B_ 256
#define T_ 512
#define D_ 64
#define H_ 512
#define SLICE 32
#define NSLICE (H_ / SLICE)   // 16 hidden-column slices
#define NTILE  (B_ / 16)      // 16 batch tiles of 16 rows (WMMA M)
#define GROUP_WGS NSLICE      // WGs cooperating on one batch tile

typedef __attribute__((ext_vector_type(16))) _Float16 v16h;
typedef __attribute__((ext_vector_type(8)))  _Float16 v8h;
typedef __attribute__((ext_vector_type(8)))  float    v8f;

union FragU { struct { v8h lo, hi; } p; v16h v; };

__device__ __forceinline__ v8f wmma_f16(v16h a, v16h b, v8f c) {
  // D = A(16x32 f16) * B(32x16 f16) + C(16x16 f32)
  return __builtin_amdgcn_wmma_f32_16x16x32_f16(false, a, false, b, (short)0, c,
                                                false, false);
}

// A fragment 16x32 f16 (ISA 7.12.2): lane m = lane&15, K runs at
// k0 + 8*(lane>=16) and k0 + 16 + 8*(lane>=16). rowPtr -> [row][0], K contiguous.
__device__ __forceinline__ v16h loadA(const _Float16* rowPtr, int k0, int o) {
  FragU u;
  u.p.lo = *(const v8h*)(rowPtr + k0 + o);
  u.p.hi = *(const v8h*)(rowPtr + k0 + 16 + o);
  return u.v;
}
// B fragment 32x16 f16: lane n = lane&15, K runs at k0 + 8*(lane>=16) and k0 + 16 + 8*(lane>=16).
// colPtr -> W[n][0] with K contiguous (W stored [n][k], i.e. B = W^T).
__device__ __forceinline__ v16h loadB(const _Float16* colPtr, int k0, int half) {
  FragU u;
  const _Float16* p = colPtr + k0 + half * 8;
  u.p.lo = *(const v8h*)(p);
  u.p.hi = *(const v8h*)(p + 16);
  return u.v;
}

// Sense-free monotonic barrier among the GROUP_WGS workgroups of one batch tile.
__device__ __forceinline__ void group_barrier(unsigned* cnt, unsigned target) {
  __threadfence();          // release this thread's global stores (agent scope)
  __syncthreads();
  if (threadIdx.x == 0) {
    __hip_atomic_fetch_add(cnt, 1u, __ATOMIC_RELAXED, __HIP_MEMORY_SCOPE_AGENT);
    while (__hip_atomic_load(cnt, __ATOMIC_RELAXED, __HIP_MEMORY_SCOPE_AGENT) < target)
      __builtin_amdgcn_s_sleep(1);
  }
  __syncthreads();
  __threadfence();          // acquire: invalidate stale lines before reading peers' h
}

__global__ __launch_bounds__(64) void rnn2_fused(
    const _Float16* __restrict__ x16,
    const _Float16* __restrict__ wih0,
    const _Float16* __restrict__ whh0,
    const _Float16* __restrict__ wih1,
    const _Float16* __restrict__ whh1,
    const float* __restrict__ bsum0,
    const float* __restrict__ bsum1,
    const float* __restrict__ wfc,
    _Float16* __restrict__ h1buf,   // 2 x B x H, double buffered
    _Float16* __restrict__ h2buf,   // 2 x B x H, double buffered
    unsigned* __restrict__ cnt,     // NTILE barrier counters
    float* __restrict__ out)        // B, pre-initialized to b_fc
{
  extern __shared__ __align__(16) _Float16 lds[];
  const int tile  = (int)blockIdx.x / NSLICE;   // batch tile
  const int slice = (int)blockIdx.x % NSLICE;   // hidden-column slice
  const int wid   = threadIdx.x >> 5;           // wave in WG (0..1)
  const int lane  = threadIdx.x & 31;
  const int half  = lane >> 4;
  const int lm    = lane & 15;
  const int tb    = tile * 16;                  // first batch row of tile
  const int col   = slice * SLICE + wid * 16 + lm;  // this lane's N column
  const int o     = half * 8;

  _Float16* ldsWih0 = lds;                      // 32 x 64
  _Float16* ldsWhh0 = ldsWih0 + SLICE * D_;     // 32 x 512
  _Float16* ldsWih1 = ldsWhh0 + SLICE * H_;     // 32 x 512
  _Float16* ldsWhh1 = ldsWih1 + SLICE * H_;     // 32 x 512  (total 100 KB)

  { // stage this slice's weights into LDS once (rows are a contiguous block)
    const uint4* s0 = (const uint4*)(wih0 + (size_t)slice * SLICE * D_);
    const uint4* s1 = (const uint4*)(whh0 + (size_t)slice * SLICE * H_);
    const uint4* s2 = (const uint4*)(wih1 + (size_t)slice * SLICE * H_);
    const uint4* s3 = (const uint4*)(whh1 + (size_t)slice * SLICE * H_);
    uint4* d0 = (uint4*)ldsWih0; uint4* d1 = (uint4*)ldsWhh0;
    uint4* d2 = (uint4*)ldsWih1; uint4* d3 = (uint4*)ldsWhh1;
    for (int i = threadIdx.x; i < SLICE * D_ / 8; i += 64) d0[i] = s0[i];
    for (int i = threadIdx.x; i < SLICE * H_ / 8; i += 64) d1[i] = s1[i];
    for (int i = threadIdx.x; i < SLICE * H_ / 8; i += 64) d2[i] = s2[i];
    for (int i = threadIdx.x; i < SLICE * H_ / 8; i += 64) d3[i] = s3[i];
  }
  __syncthreads();

  const _Float16* bW0 = ldsWih0 + (wid * 16 + lm) * D_;
  const _Float16* bH0 = ldsWhh0 + (wid * 16 + lm) * H_;
  const _Float16* bW1 = ldsWih1 + (wid * 16 + lm) * H_;
  const _Float16* bH1 = ldsWhh1 + (wid * 16 + lm) * H_;

  const float bias0 = bsum0[col];
  const float bias1 = bsum1[col];
  const float wfcv  = wfc[col];

  unsigned* myCnt = cnt + tile;
  unsigned target = 0;
  const _Float16* xrow = x16 + (size_t)(tb + lm) * T_ * D_;

  for (int t = 0; t < T_; ++t) {
    const int cb = t & 1, nb = cb ^ 1;

    // ---------- layer 1: h1_t = tanh(x_t·Wih0^T + h1_{t-1}·Whh0^T + b) ----------
    v8f acc = {};
    const _Float16* xr = xrow + (size_t)t * D_;
    #pragma unroll
    for (int k0 = 0; k0 < D_; k0 += 32)
      acc = wmma_f16(loadA(xr, k0, o), loadB(bW0, k0, half), acc);
    if (t + 1 < T_) __builtin_prefetch(xr + D_, 0, 1);   // global_prefetch_b8
    const _Float16* h1p = h1buf + ((size_t)cb * B_ + tb + lm) * H_;
    #pragma unroll 4
    for (int k0 = 0; k0 < H_; k0 += 32)
      acc = wmma_f16(loadA(h1p, k0, o), loadB(bH0, k0, half), acc);

    float hv[8];
    #pragma unroll
    for (int r = 0; r < 8; ++r) hv[r] = tanhf(acc[r] + bias0);
    _Float16* h1n = h1buf + (size_t)nb * B_ * H_;
    #pragma unroll
    for (int r = 0; r < 8; ++r)       // C layout: lane n=lm, rows M=r+8*half
      h1n[(size_t)(tb + r + half * 8) * H_ + col] = (_Float16)hv[r];

    target += GROUP_WGS;
    group_barrier(myCnt, target);     // full h1_t visible to all 16 slice-WGs

    // ---------- layer 2: h2_t = tanh(h1_t·Wih1^T + h2_{t-1}·Whh1^T + b) --------
    v8f acc2 = {};
    const _Float16* h1r = h1buf + ((size_t)nb * B_ + tb + lm) * H_;
    const _Float16* h2r = h2buf + ((size_t)cb * B_ + tb + lm) * H_;
    #pragma unroll 4
    for (int k0 = 0; k0 < H_; k0 += 32)
      acc2 = wmma_f16(loadA(h1r, k0, o), loadB(bW1, k0, half), acc2);
    #pragma unroll 4
    for (int k0 = 0; k0 < H_; k0 += 32)
      acc2 = wmma_f16(loadA(h2r, k0, o), loadB(bH1, k0, half), acc2);

    #pragma unroll
    for (int r = 0; r < 8; ++r) hv[r] = tanhf(acc2[r] + bias1);
    _Float16* h2n = h2buf + (size_t)nb * B_ * H_;
    #pragma unroll
    for (int r = 0; r < 8; ++r)
      h2n[(size_t)(tb + r + half * 8) * H_ + col] = (_Float16)hv[r];

    if (t == T_ - 1) {                // fused FC head on the final hidden state
      #pragma unroll
      for (int r = 0; r < 8; ++r)
        atomicAdd(&out[tb + r + half * 8], hv[r] * wfcv);
    }

    target += GROUP_WGS;
    group_barrier(myCnt, target);     // h2_t complete; safe to recycle buffers
  }
}

// ---------------- prep kernels (run every launch; graph-replay safe) ----------
__global__ void k_cvt(const float* __restrict__ in, _Float16* __restrict__ out, int n) {
  int i = blockIdx.x * blockDim.x + threadIdx.x;
  if (i < n) out[i] = (_Float16)in[i];
}
__global__ void k_bias(const float* __restrict__ a, const float* __restrict__ b,
                       float* __restrict__ o, int n) {
  int i = blockIdx.x * blockDim.x + threadIdx.x;
  if (i < n) o[i] = a[i] + b[i];
}
__global__ void k_init(_Float16* __restrict__ h1, _Float16* __restrict__ h2,
                       unsigned* __restrict__ cnt, float* __restrict__ out,
                       const float* __restrict__ bfc, int nh) {
  int i = blockIdx.x * blockDim.x + threadIdx.x;
  if (i < nh) { h1[i] = (_Float16)0.f; h2[i] = (_Float16)0.f; }
  if (i < NTILE) cnt[i] = 0u;
  if (i < B_) out[i] = bfc[0];
}

extern "C" void kernel_launch(void* const* d_in, const int* in_sizes, int n_in,
                              void* d_out, int out_size, void* d_ws, size_t ws_size,
                              hipStream_t stream) {
  (void)in_sizes; (void)n_in; (void)out_size; (void)ws_size;
  const float* x    = (const float*)d_in[0];
  const float* Wih0 = (const float*)d_in[1];
  const float* Whh0 = (const float*)d_in[2];
  const float* bih0 = (const float*)d_in[3];
  const float* bhh0 = (const float*)d_in[4];
  const float* Wih1 = (const float*)d_in[5];
  const float* Whh1 = (const float*)d_in[6];
  const float* bih1 = (const float*)d_in[7];
  const float* bhh1 = (const float*)d_in[8];
  const float* Wfc  = (const float*)d_in[9];
  const float* bfc  = (const float*)d_in[10];
  float* out = (float*)d_out;

  // workspace layout (~19.4 MB, 16B-aligned slabs)
  _Float16* x16   = (_Float16*)d_ws;            // B*T*D
  _Float16* wih0h = x16   + (size_t)B_ * T_ * D_;
  _Float16* whh0h = wih0h + (size_t)H_ * D_;
  _Float16* wih1h = whh0h + (size_t)H_ * H_;
  _Float16* whh1h = wih1h + (size_t)H_ * H_;
  _Float16* h1buf = whh1h + (size_t)H_ * H_;    // 2*B*H
  _Float16* h2buf = h1buf + (size_t)2 * B_ * H_;
  float*    bsum0 = (float*)(h2buf + (size_t)2 * B_ * H_);
  float*    bsum1 = bsum0 + H_;
  unsigned* cnt   = (unsigned*)(bsum1 + H_);

  const int nx = B_ * T_ * D_;
  k_cvt<<<(nx + 255) / 256, 256, 0, stream>>>(x, x16, nx);
  k_cvt<<<(H_ * D_ + 255) / 256, 256, 0, stream>>>(Wih0, wih0h, H_ * D_);
  k_cvt<<<(H_ * H_ + 255) / 256, 256, 0, stream>>>(Whh0, whh0h, H_ * H_);
  k_cvt<<<(H_ * H_ + 255) / 256, 256, 0, stream>>>(Wih1, wih1h, H_ * H_);
  k_cvt<<<(H_ * H_ + 255) / 256, 256, 0, stream>>>(Whh1, whh1h, H_ * H_);
  k_bias<<<(H_ + 255) / 256, 256, 0, stream>>>(bih0, bhh0, bsum0, H_);
  k_bias<<<(H_ + 255) / 256, 256, 0, stream>>>(bih1, bhh1, bsum1, H_);
  const int nh = 2 * B_ * H_;
  k_init<<<(nh + 255) / 256, 256, 0, stream>>>(h1buf, h2buf, cnt, out, bfc, nh);

  const size_t smem = (size_t)(SLICE * D_ + 3 * SLICE * H_) * sizeof(_Float16); // 100 KB
  rnn2_fused<<<NTILE * NSLICE, 64, smem, stream>>>(
      x16, wih0h, whh0h, wih1h, whh1h, bsum0, bsum1, Wfc, h1buf, h2buf, cnt, out);
}
